// MultiHeadSelfAttention_85203561218186
// MI455X (gfx1250) — matrix operand order repaired and checked by reading, not compiled, not run on hardware
//
#include <hip/hip_runtime.h>
#include <hip/hip_bf16.h>

typedef _Float16 v16h __attribute__((ext_vector_type(16)));
typedef _Float16 v8h  __attribute__((ext_vector_type(8)));
typedef float    v8f  __attribute__((ext_vector_type(8)));

#define DIMN   1024
#define SEQ    2048
#define NBATCH 2
#define TOK    (NBATCH*SEQ)
#define NHEADS 16
#define HDIM   64

// ---- fragment loaders (16-bit WMMA layouts, wave32) -------------------------
// A 16x32 (MxK): lane m = lane&15; lanes0-15: elems0-7=K+0..7, elems8-15=K+16..23
//                lanes16-31: elems0-7=K+8..15, elems8-15=K+24..31
// caller passes p already offset by (hi ? 8 : 0)
static __device__ inline v16h load_frag_a(const _Float16* p) {
  v8h lo = *(const v8h*)p;
  v8h hi = *(const v8h*)(p + 16);
  v16h r;
#pragma unroll
  for (int i = 0; i < 8; ++i) { r[i] = lo[i]; r[i + 8] = hi[i]; }
  return r;
}
// B 32x16 (KxN): lane n = lane&15; lanes0-15: elems0-7=K+0..7, elems8-15=K+16..23
//                lanes16-31: elems0-7=K+8..15, elems8-15=K+24..31
// caller passes p already offset by (hi ? 8 : 0).
static __device__ inline v16h load_frag_b(const _Float16* p) {
  v8h lo = *(const v8h*)p;
  v8h hi = *(const v8h*)(p + 16);
  v16h r;
#pragma unroll
  for (int i = 0; i < 8; ++i) { r[i] = lo[i]; r[i + 8] = hi[i]; }
  return r;
}
static __device__ inline v8f zero8() {
  v8f z;
#pragma unroll
  for (int i = 0; i < 8; ++i) z[i] = 0.f;
  return z;
}
static __device__ inline v8f wmma_f16(v16h a, v16h b, v8f c) {
  return __builtin_amdgcn_wmma_f32_16x16x32_f16(false, a, false, b, (short)0, c,
                                                false, false);
}

// ---- RMSNorm (f32 in -> f16 out) -------------------------------------------
__global__ __launch_bounds__(256) void rmsnorm_f16(const float* __restrict__ x,
                                                   const float* __restrict__ w,
                                                   _Float16* __restrict__ xn) {
  const int t = blockIdx.x;
  const float* row = x + (size_t)t * DIMN;
  __shared__ float red[256];
  float s = 0.f;
  for (int c = threadIdx.x; c < DIMN; c += 256) { float v = row[c]; s += v * v; }
  red[threadIdx.x] = s;
  __syncthreads();
  for (int off = 128; off > 0; off >>= 1) {
    if (threadIdx.x < off) red[threadIdx.x] += red[threadIdx.x + off];
    __syncthreads();
  }
  const float rinv = rsqrtf(red[0] * (1.f / DIMN) + 1e-5f);
  _Float16* orow = xn + (size_t)t * DIMN;
  for (int c = threadIdx.x; c < DIMN; c += 256)
    orow[c] = (_Float16)(row[c] * rinv * w[c]);
}

// ---- weight convert + transpose: Wt[n][k] = (f16) W[k][n] ------------------
__global__ __launch_bounds__(256) void conv_w_f16t(const float* __restrict__ W,
                                                   _Float16* __restrict__ Wt) {
  int idx = blockIdx.x * 256 + threadIdx.x;
  if (idx >= DIMN * DIMN) return;
  int n = idx / DIMN, k = idx % DIMN;
  Wt[(size_t)n * DIMN + k] = (_Float16)W[(size_t)k * DIMN + n];
}

// ---- generic WMMA GEMM: out[M x 1024] = A[M x 1024] @ Bt^T -----------------
// Each wave computes a 32x64 output tile: 2 A-fragments x 4 B-fragments
// (8 WMMAs per 32-K step; 12 b128 loads -> 1.5 loads/WMMA).
// mode 0: f16 row-major out.  mode 1: f16 out transposed to [b,h,d,seq].
// mode 2: f32 row-major out (final projection into d_out).
__global__ __launch_bounds__(32) void gemm_wmma(const _Float16* __restrict__ A,
                                                const _Float16* __restrict__ Bt,
                                                void* __restrict__ out,
                                                int mode) {
  const int lane = threadIdx.x;
  const int col  = lane & 15;
  const int hi   = (lane >> 4) & 1;
  const int m0   = blockIdx.x * 32;
  const int n0   = blockIdx.y * 64;
  const _Float16* arow0 = A + (size_t)(m0 + col) * DIMN;
  const _Float16* arow1 = A + (size_t)(m0 + 16 + col) * DIMN;
  v8f c[2][4];
#pragma unroll
  for (int mi = 0; mi < 2; ++mi)
#pragma unroll
    for (int j = 0; j < 4; ++j) c[mi][j] = zero8();

  for (int k = 0; k < DIMN; k += 32) {
    v16h a0 = load_frag_a(arow0 + k + hi * 8);
    v16h a1 = load_frag_a(arow1 + k + hi * 8);
#pragma unroll
    for (int j = 0; j < 4; ++j) {
      v16h b = load_frag_b(Bt + (size_t)(n0 + j * 16 + col) * DIMN + k + hi * 8);
      c[0][j] = wmma_f16(a0, b, c[0][j]);
      c[1][j] = wmma_f16(a1, b, c[1][j]);
    }
  }
  if (mode == 0) {
    _Float16* O = (_Float16*)out;
#pragma unroll
    for (int mi = 0; mi < 2; ++mi)
#pragma unroll
      for (int j = 0; j < 4; ++j) {
        const int n = n0 + j * 16 + col;
#pragma unroll
        for (int r = 0; r < 8; ++r) {
          const int m = m0 + mi * 16 + r + hi * 8;
          O[(size_t)m * DIMN + n] = (_Float16)c[mi][j][r];
        }
      }
  } else if (mode == 1) {
    _Float16* O = (_Float16*)out;
#pragma unroll
    for (int mi = 0; mi < 2; ++mi)
#pragma unroll
      for (int j = 0; j < 4; ++j) {
        const int n = n0 + j * 16 + col;
        const int hh = n >> 6, d = n & 63;
#pragma unroll
        for (int r = 0; r < 8; ++r) {
          const int m = m0 + mi * 16 + r + hi * 8;
          const int bb = m >> 11, t = m & (SEQ - 1);
          O[((size_t)(bb * NHEADS + hh) * HDIM + d) * SEQ + t] =
              (_Float16)c[mi][j][r];
        }
      }
  } else {
    float* O = (float*)out;
#pragma unroll
    for (int mi = 0; mi < 2; ++mi)
#pragma unroll
      for (int j = 0; j < 4; ++j) {
        const int n = n0 + j * 16 + col;
#pragma unroll
        for (int r = 0; r < 8; ++r) {
          const int m = m0 + mi * 16 + r + hi * 8;
          O[(size_t)m * DIMN + n] = c[mi][j][r];
        }
      }
  }
}

// ---- RoPE in place on f16 [TOK x 1024]; scale folded into Q ----------------
__global__ __launch_bounds__(256) void rope_inplace(_Float16* __restrict__ arr,
                                                    float scale) {
  int idx = blockIdx.x * 256 + threadIdx.x;
  if (idx >= TOK * NHEADS * (HDIM / 2)) return;
  const int i = idx & 31;
  const int h = (idx >> 5) & 15;
  const int t = idx >> 9;
  const int pos = t & (SEQ - 1);
  const float theta = __powf(1.0e6f, -(float)i * (1.f / 32.f));
  float sn, cs;
  __sincosf((float)pos * theta, &sn, &cs);
  _Float16* p = arr + (size_t)t * DIMN + h * HDIM;
  const float x1 = (float)p[i], x2 = (float)p[i + 32];
  p[i]      = (_Float16)((x1 * cs - x2 * sn) * scale);
  p[i + 32] = (_Float16)((x1 * sn + x2 * cs) * scale);
}

// ---- flash attention: one wave per (b, h, 16-query tile) -------------------
__global__ __launch_bounds__(32) void flash_attn(const _Float16* __restrict__ Q,
                                                 const _Float16* __restrict__ K,
                                                 const _Float16* __restrict__ Vt,
                                                 _Float16* __restrict__ Ao) {
  const int lane = threadIdx.x;
  const int col  = lane & 15;
  const int hi   = (lane >> 4) & 1;
  const int q0   = blockIdx.x * 16;
  const int h    = blockIdx.y;
  const int b    = blockIdx.z;
  const size_t tok0 = (size_t)b * SEQ;
  __shared__ _Float16 pbuf[16 * 32];

  const _Float16* qrow = Q + (tok0 + q0 + col) * DIMN + h * HDIM;
  const v16h aq0 = load_frag_a(qrow + 0  + hi * 8);   // d = 0..31
  const v16h aq1 = load_frag_a(qrow + 32 + hi * 8);   // d = 32..63

  v8f o[4] = {zero8(), zero8(), zero8(), zero8()};
  float m[8], l[8];
#pragma unroll
  for (int r = 0; r < 8; ++r) { m[r] = -1e30f; l[r] = 0.f; }

  for (int j = 0; j <= q0 + 15; j += 32) {
    // S = Q @ K^T for keys [j, j+32); two 16-key subtiles
    v8f s0 = zero8(), s1 = zero8();
    {
      const _Float16* k0 = K + (tok0 + j + col) * DIMN + h * HDIM + hi * 8;
      const _Float16* k1 = K + (tok0 + j + 16 + col) * DIMN + h * HDIM + hi * 8;
      v16h bk;
      bk = load_frag_b(k0);       s0 = wmma_f16(aq0, bk, s0);
      bk = load_frag_b(k0 + 32);  s0 = wmma_f16(aq1, bk, s0);
      bk = load_frag_b(k1);       s1 = wmma_f16(aq0, bk, s1);
      bk = load_frag_b(k1 + 32);  s1 = wmma_f16(aq1, bk, s1);
    }
    // causal mask + online softmax (row r lives across 16 lanes of this half)
#pragma unroll
    for (int r = 0; r < 8; ++r) {
      const int tg = q0 + r + hi * 8;
      float v0 = (j + col      <= tg) ? s0[r] : -1e30f;
      float v1 = (j + 16 + col <= tg) ? s1[r] : -1e30f;
      float mx = fmaxf(v0, v1);
      mx = fmaxf(mx, __shfl_xor(mx, 1));
      mx = fmaxf(mx, __shfl_xor(mx, 2));
      mx = fmaxf(mx, __shfl_xor(mx, 4));
      mx = fmaxf(mx, __shfl_xor(mx, 8));
      const float mn = fmaxf(m[r], mx);
      const float alpha = __expf(m[r] - mn);
      const float p0 = __expf(v0 - mn);
      const float p1 = __expf(v1 - mn);
      float rs = p0 + p1;
      rs += __shfl_xor(rs, 1);
      rs += __shfl_xor(rs, 2);
      rs += __shfl_xor(rs, 4);
      rs += __shfl_xor(rs, 8);
      l[r] = l[r] * alpha + rs;
      m[r] = mn;
      o[0][r] *= alpha; o[1][r] *= alpha; o[2][r] *= alpha; o[3][r] *= alpha;
      s0[r] = p0; s1[r] = p1;
    }
    // transpose P from C/D layout to A layout via LDS
    __syncthreads();
#pragma unroll
    for (int r = 0; r < 8; ++r) {
      pbuf[(r + hi * 8) * 32 + col]      = (_Float16)s0[r];
      pbuf[(r + hi * 8) * 32 + 16 + col] = (_Float16)s1[r];
    }
    __syncthreads();
    const v16h ap = load_frag_a(&pbuf[col * 32 + hi * 8]);
    // O += P @ V  (Vt layout [b,h,d,seq] -> contiguous along keys)
#pragma unroll
    for (int ch = 0; ch < 4; ++ch) {
      const _Float16* vr =
          Vt + ((size_t)((b * NHEADS + h) * HDIM) + ch * 16 + col) * SEQ + j + hi * 8;
      v16h bv = load_frag_b(vr);
      o[ch] = wmma_f16(ap, bv, o[ch]);
    }
    __syncthreads();
  }
  // epilogue: divide by row sum, store f16 row-major [TOK x 1024]
#pragma unroll
  for (int ch = 0; ch < 4; ++ch) {
#pragma unroll
    for (int r = 0; r < 8; ++r) {
      const int tg = q0 + r + hi * 8;
      Ao[(tok0 + tg) * DIMN + h * HDIM + ch * 16 + col] = (_Float16)(o[ch][r] / l[r]);
    }
  }
}

// ---- host-side orchestration ------------------------------------------------
extern "C" void kernel_launch(void* const* d_in, const int* in_sizes, int n_in,
                              void* d_out, int out_size, void* d_ws, size_t ws_size,
                              hipStream_t stream) {
  (void)in_sizes; (void)n_in; (void)out_size; (void)ws_size;
  const float* x  = (const float*)d_in[0];
  const float* wn = (const float*)d_in[1];
  const float* Wq = (const float*)d_in[2];
  const float* Wk = (const float*)d_in[3];
  const float* Wv = (const float*)d_in[4];
  const float* Wo = (const float*)d_in[5];

  char* ws = (char*)d_ws;
  const size_t actB = (size_t)TOK * DIMN * sizeof(_Float16);  // 8 MB
  const size_t wB   = (size_t)DIMN * DIMN * sizeof(_Float16); // 2 MB
  _Float16* xn  = (_Float16*)(ws);
  _Float16* wqh = (_Float16*)(ws + actB);
  _Float16* wkh = (_Float16*)(ws + actB + 1 * wB);
  _Float16* wvh = (_Float16*)(ws + actB + 2 * wB);
  _Float16* woh = (_Float16*)(ws + actB + 3 * wB);
  _Float16* Qh  = (_Float16*)(ws + 1 * actB + 4 * wB);
  _Float16* Kh  = (_Float16*)(ws + 2 * actB + 4 * wB);
  _Float16* Vt  = (_Float16*)(ws + 3 * actB + 4 * wB);
  _Float16* Ah  = (_Float16*)(ws + 4 * actB + 4 * wB);

  rmsnorm_f16<<<TOK, 256, 0, stream>>>(x, wn, xn);
  const int wgrid = (DIMN * DIMN + 255) / 256;
  conv_w_f16t<<<wgrid, 256, 0, stream>>>(Wq, wqh);
  conv_w_f16t<<<wgrid, 256, 0, stream>>>(Wk, wkh);
  conv_w_f16t<<<wgrid, 256, 0, stream>>>(Wv, wvh);
  conv_w_f16t<<<wgrid, 256, 0, stream>>>(Wo, woh);

  dim3 ggrid(TOK / 32, DIMN / 64);
  gemm_wmma<<<ggrid, 32, 0, stream>>>(xn, wqh, Qh, 0);
  gemm_wmma<<<ggrid, 32, 0, stream>>>(xn, wkh, Kh, 0);
  gemm_wmma<<<ggrid, 32, 0, stream>>>(xn, wvh, Vt, 1);

  const int rgrid = (TOK * NHEADS * (HDIM / 2) + 255) / 256;
  rope_inplace<<<rgrid, 256, 0, stream>>>(Qh, 0.125f);  // fold 1/sqrt(64) into Q
  rope_inplace<<<rgrid, 256, 0, stream>>>(Kh, 1.0f);

  flash_attn<<<dim3(SEQ / 16, NHEADS, NBATCH), 32, 0, stream>>>(Qh, Kh, Vt, Ah);

  gemm_wmma<<<ggrid, 32, 0, stream>>>(Ah, woh, d_out, 2);
}
